// MultiHeadAttentionSigmoid_75788992905596
// MI455X (gfx1250) — matrix operand order repaired and checked by reading, not compiled, not run on hardware
//
#include <hip/hip_runtime.h>
#include <hip/hip_bf16.h>

typedef __attribute__((ext_vector_type(16))) _Float16 v16h;
typedef __attribute__((ext_vector_type(8)))  float    v8f;

#define D_MODEL 1024
#define NHEADS  16
#define HDIM    64
#define SEQ     2048
#define BATCH   2
#define MTOT    (BATCH*SEQ)          // 4096 token rows
#define SLAB    64                   // keys staged per LDS slab

// q is pre-scaled by (1/sqrt(HDIM)) * log2(e) so sigmoid is rcp(1 + exp2(-s))
#define QSCALE  0.18033688f

// ---------------------------------------------------------------------------
// Fragment loaders (wave32 WMMA 16x16x32 f16 layouts, CDNA5 ISA)
// A 16x32: lane m=l&15; elems 0..7 = K(hi*8..hi*8+7), elems 8..15 = +16
// B 32x16: lane n=l&15; elems 0..7 = K(hi*8..hi*8+7), elems 8..15 = +16
// ---------------------------------------------------------------------------
__device__ __forceinline__ v16h load_fragA(const _Float16* p) {
    v16h r;
    *(uint4*)&r       = *(const uint4*)(p);
    *((uint4*)&r + 1) = *(const uint4*)(p + 16);
    return r;
}
__device__ __forceinline__ v16h load_fragB(const _Float16* p) {
    v16h r;
    *(uint4*)&r       = *(const uint4*)(p);
    *((uint4*)&r + 1) = *(const uint4*)(p + 16);
    return r;
}

// ---------------------------------------------------------------------------
// Tensor Data Mover staging (gfx1250).  D# layout per CDNA5 ISA
// ---------------------------------------------------------------------------
#if __has_builtin(__builtin_amdgcn_tensor_load_to_lds)
#define HAS_TDM 1
typedef unsigned __attribute__((ext_vector_type(4))) u32x4;
typedef int      __attribute__((ext_vector_type(4))) i32x4;
typedef int      __attribute__((ext_vector_type(8))) i32x8;

__device__ __forceinline__ void tdm_load_2d(unsigned lds_byte, const void* gptr,
                                            unsigned td0, unsigned td1,
                                            unsigned tile0, unsigned tile1,
                                            unsigned stride0) {
    unsigned long long ga = (unsigned long long)(uintptr_t)gptr;
    u32x4 g0;
    g0[0] = 1u;                                  // count=1 (valid user D#)
    g0[1] = lds_byte;                            // lds_addr (bytes)
    g0[2] = (unsigned)ga;                        // global_addr lo
    g0[3] = (unsigned)(ga >> 32) | (2u << 30);   // global_addr hi | type=2
    i32x8 g1;
    g1[0] = 0x00010000;                          // data_size = 2 bytes
    g1[1] = (int)(td0 << 16);                    // tensor_dim0 lo16
    g1[2] = (int)((td0 >> 16) | (td1 << 16));    // dim0 hi16 | dim1 lo16
    g1[3] = (int)((td1 >> 16) | (tile0 << 16));  // dim1 hi16 | tile_dim0
    g1[4] = (int)(tile1 & 0xffff);               // tile_dim1 (tile_dim2 = 0)
    g1[5] = (int)stride0;                        // tensor_dim0_stride lo32
    g1[6] = 0;
    g1[7] = 0;
    i32x4 z4 = {0, 0, 0, 0};
#if __clang_major__ >= 23
    i32x8 z8 = {0, 0, 0, 0, 0, 0, 0, 0};
    __builtin_amdgcn_tensor_load_to_lds(g0, g1, z4, z4, z8, 0);
#else
    __builtin_amdgcn_tensor_load_to_lds(g0, g1, z4, z4, 0);
#endif
}
#endif

// ---------------------------------------------------------------------------
// f32 -> f16 conversion (weights)
// ---------------------------------------------------------------------------
__global__ void f32_to_f16_kernel(const float* __restrict__ in,
                                  _Float16* __restrict__ out, int n) {
    int i = blockIdx.x * blockDim.x + threadIdx.x;
    if (i < n) out[i] = (_Float16)in[i];
}

// ---------------------------------------------------------------------------
// RMSNorm: one block per token row; emits q-normed, k-normed, raw (all f16)
// ---------------------------------------------------------------------------
__global__ void __launch_bounds__(256)
rmsnorm_prep_kernel(const float* __restrict__ x,
                    const float* __restrict__ qn, const float* __restrict__ kn,
                    _Float16* __restrict__ xq, _Float16* __restrict__ xk,
                    _Float16* __restrict__ xv) {
    __shared__ float red[256];
    const int row = blockIdx.x;
    const float* xr = x + (size_t)row * D_MODEL;
    float s = 0.f;
    for (int i = threadIdx.x; i < D_MODEL; i += 256) { float v = xr[i]; s += v * v; }
    red[threadIdx.x] = s;
    __syncthreads();
    for (int off = 128; off > 0; off >>= 1) {
        if (threadIdx.x < off) red[threadIdx.x] += red[threadIdx.x + off];
        __syncthreads();
    }
    const float rstd = rsqrtf(red[0] * (1.0f / D_MODEL) + 1e-8f);
    for (int i = threadIdx.x; i < D_MODEL; i += 256) {
        float v  = xr[i];
        float nv = v * rstd;
        size_t o = (size_t)row * D_MODEL + i;
        xq[o] = (_Float16)(nv * qn[i]);
        xk[o] = (_Float16)(nv * kn[i]);
        xv[o] = (_Float16)v;
    }
}

// ---------------------------------------------------------------------------
// GEMM: C[M,1024] = A[M,1024] @ W[1024,1024]^T + bias   (TN, K-contig both)
// 256 thr = 8 waves (2 Mwave x 4 Nwave), wave tile 64x64 = 16 wmma accums
// grid = (N/256, M/128).  Output mode selects layout in the scalar-store
// epilogue (free: address math only):
//   0: f16 [m, 1024]          1: f32 [m, 1024] (final output)
//   2: f16 [b,h,l,d] (q/k)    3: f16 [b,h,d,l] (v transposed)
// ---------------------------------------------------------------------------
template <int MODE>
__global__ void __launch_bounds__(256)
gemm_bias_kernel(const _Float16* __restrict__ A, const _Float16* __restrict__ W,
                 const float* __restrict__ bias, void* __restrict__ Cout) {
    const int lane  = threadIdx.x & 31;
    const int wid   = threadIdx.x >> 5;
    const int row16 = lane & 15;
    const int hi    = lane >> 4;
    const int m0    = blockIdx.y * 128 + (wid >> 2) * 64;
    const int n0    = blockIdx.x * 256 + (wid & 3) * 64;

    v8f zero = {};
    v8f acc[4][4];
    for (int i = 0; i < 4; i++)
        for (int j = 0; j < 4; j++) acc[i][j] = zero;

    for (int k = 0; k < D_MODEL; k += 32) {
        // hint next K slab toward the caches (global_prefetch_b8)
        __builtin_prefetch(A + (size_t)(m0 + row16) * D_MODEL + k + 256, 0, 0);
        __builtin_prefetch(W + (size_t)(n0 + row16) * D_MODEL + k + 256, 0, 0);
        v16h af[4], bf[4];
        for (int i = 0; i < 4; i++)
            af[i] = load_fragA(A + (size_t)(m0 + i * 16 + row16) * D_MODEL + k + hi * 8);
        for (int j = 0; j < 4; j++)
            bf[j] = load_fragB(W + (size_t)(n0 + j * 16 + row16) * D_MODEL + k + hi * 8);
        for (int i = 0; i < 4; i++)
            for (int j = 0; j < 4; j++)
                acc[i][j] = __builtin_amdgcn_wmma_f32_16x16x32_f16(
                    false, af[i], false, bf[j], (short)0, acc[i][j], false, false);
    }

    for (int i = 0; i < 4; i++) {
        const int mbase = m0 + i * 16 + hi * 8;
        for (int j = 0; j < 4; j++) {
            const int ncol = n0 + j * 16 + row16;
            const float bv = bias[ncol];
            for (int r = 0; r < 8; r++) {
                float v = acc[i][j][r] + bv;
                const int m = mbase + r;
                if (MODE == 1) {
                    ((float*)Cout)[(size_t)m * D_MODEL + ncol] = v;
                } else if (MODE == 0) {
                    ((_Float16*)Cout)[(size_t)m * D_MODEL + ncol] = (_Float16)v;
                } else {
                    const int b = m >> 11, l = m & (SEQ - 1);
                    const int h = ncol >> 6, d = ncol & (HDIM - 1);
                    size_t o;
                    if (MODE == 2)
                        o = (((size_t)(b * NHEADS + h)) * SEQ + l) * HDIM + d;
                    else
                        o = (((size_t)(b * NHEADS + h)) * HDIM + d) * SEQ + l;
                    ((_Float16*)Cout)[o] = (_Float16)v;
                }
            }
        }
    }
}

// ---------------------------------------------------------------------------
// RoPE on [b,h,l,d] layout (coalesced, in/out same layout).  q additionally
// folds in QSCALE so attention scores feed exp2 directly.
// ---------------------------------------------------------------------------
__global__ void rope_kernel(const _Float16* __restrict__ qp,
                            const _Float16* __restrict__ kp,
                            _Float16* __restrict__ qr,
                            _Float16* __restrict__ kr) {
    int idx = blockIdx.x * blockDim.x + threadIdx.x;
    if (idx >= BATCH * NHEADS * SEQ * (HDIM / 2)) return;
    const int i = idx & 31;
    const int l = (idx >> 5) & (SEQ - 1);
    const int h = (idx >> 16) & (NHEADS - 1);
    const int b = idx >> 20;
    const int d1 = 2 * i, d2 = 2 * i + 1;
    const float a1 = (float)l * powf(10000.0f, -(float)(d1 & 31) * (1.0f / 32.0f));
    const float a2 = (float)l * powf(10000.0f, -(float)(d2 & 31) * (1.0f / 32.0f));
    const float c1 = cosf(a1), s1 = sinf(a1);
    const float c2 = cosf(a2), s2 = sinf(a2);
    const size_t base = (((size_t)(b * NHEADS + h)) * SEQ + l) * HDIM;
    {
        float x1 = (float)qp[base + d1], x2 = (float)qp[base + d2];
        qr[base + d1] = (_Float16)((x1 * c1 - x2 * s1) * QSCALE);
        qr[base + d2] = (_Float16)((x2 * c2 + x1 * s2) * QSCALE);
    }
    {
        float x1 = (float)kp[base + d1], x2 = (float)kp[base + d2];
        kr[base + d1] = (_Float16)(x1 * c1 - x2 * s1);
        kr[base + d2] = (_Float16)(x2 * c2 + x1 * s2);
    }
}

// ---------------------------------------------------------------------------
// Sigmoid attention, flash-style single pass.
// grid = (B*H, SEQ/128), block = 128 (4 waves); each wave owns 32 q rows
// (2 m-tiles) so every staged K/V fragment feeds 2 WMMAs.  All 4 waves share
// (b,h), so 64-key K / V^T slabs are staged into double-buffered LDS by the
// Tensor Data Mover (wave 0 issues; s_wait_tensorcnt + barrier).  Per 32-key
// step: 8 WMMAs for S, sigmoid = v_exp2+v_rcp, LDS restage of W into
// A-layout, 8 WMMAs for O.  Renorm by (rowsum + eps) at the end.
// ---------------------------------------------------------------------------
__global__ void __launch_bounds__(128)
attn_sigmoid_kernel(const _Float16* __restrict__ qr,
                    const _Float16* __restrict__ kr,
                    const _Float16* __restrict__ vt,
                    _Float16* __restrict__ attn) {
    __shared__ _Float16 kbuf[2][SLAB * HDIM];     // [key][d]
    __shared__ _Float16 vbuf[2][SLAB * HDIM];     // [d][key]
    __shared__ _Float16 wstage[4][2 * 16 * 32];   // per-wave, 2 m-tiles
    const int lane  = threadIdx.x & 31;
    const int wid   = threadIdx.x >> 5;
    const int row16 = lane & 15;
    const int hi    = lane >> 4;
    const int bh    = blockIdx.x;
    const int b     = bh >> 4, h = bh & (NHEADS - 1);
    const int l0    = blockIdx.y * 128 + wid * 32;

    const _Float16* qg = qr + (size_t)bh * SEQ * HDIM;
    const _Float16* kg = kr + (size_t)bh * SEQ * HDIM;
    const _Float16* vg = vt + (size_t)bh * HDIM * SEQ;
    _Float16* ws = wstage[wid];

    auto stage = [&](int buf, int k0) {
#ifdef HAS_TDM
        if (wid == 0) {
            // K slab: rows = keys (tile1=64), row width = HDIM contiguous
            tdm_load_2d((unsigned)(uintptr_t)kbuf[buf], kg + (size_t)k0 * HDIM,
                        HDIM, SEQ, HDIM, SLAB, HDIM);
            // V^T slab: rows = d (tile1=64), 64 contiguous keys per row
            tdm_load_2d((unsigned)(uintptr_t)vbuf[buf], vg + k0,
                        SEQ, HDIM, SLAB, HDIM, SEQ);
        }
#else
        for (int i = threadIdx.x; i < SLAB * HDIM / 8; i += 128) {
            ((uint4*)kbuf[buf])[i] =
                ((const uint4*)(kg + (size_t)k0 * HDIM))[i];
            int d = i >> 3, c = i & 7;
            ((uint4*)vbuf[buf])[i] =
                *(const uint4*)(vg + (size_t)d * SEQ + k0 + c * 8);
        }
#endif
    };
    auto wait_stage = [&]() {
#ifdef HAS_TDM
        if (wid == 0) __builtin_amdgcn_s_wait_tensorcnt((short)0);
#endif
        __syncthreads();
    };

    // q A-fragments: 2 m-tiles x (64 d -> 2 K-chunks of 32)
    v16h qf[2][2];
    for (int mt = 0; mt < 2; mt++) {
        const _Float16* qp = qg + (size_t)(l0 + mt * 16 + row16) * HDIM;
        qf[mt][0] = load_fragA(qp + 0  + hi * 8);
        qf[mt][1] = load_fragA(qp + 32 + hi * 8);
    }

    v8f zero = {};
    v8f oacc[2][4];
    float rowsum[2][8];
    for (int mt = 0; mt < 2; mt++) {
        for (int j = 0; j < 4; j++) oacc[mt][j] = zero;
        for (int r = 0; r < 8; r++) rowsum[mt][r] = 0.f;
    }

    stage(0, 0);
    wait_stage();

    for (int s = 0; s < SEQ / SLAB; ++s) {
        if (s + 1 < SEQ / SLAB) stage((s + 1) & 1, (s + 1) * SLAB);
        const _Float16* kb = kbuf[s & 1];
        const _Float16* vb = vbuf[s & 1];

        for (int kk = 0; kk < SLAB; kk += 32) {
            // gather all K fragments first, then burst the score WMMAs
            v16h kf[2][2];
            for (int t = 0; t < 2; t++) {
                const _Float16* kp = kb + (size_t)(kk + t * 16 + row16) * HDIM;
                kf[t][0] = load_fragB(kp + 0  + hi * 8);
                kf[t][1] = load_fragB(kp + 32 + hi * 8);
            }
            v8f sc[2][2];
            for (int mt = 0; mt < 2; mt++)
                for (int t = 0; t < 2; t++) {
                    v8f a = __builtin_amdgcn_wmma_f32_16x16x32_f16(
                        false, qf[mt][0], false, kf[t][0], (short)0, zero, false, false);
                    sc[mt][t] = __builtin_amdgcn_wmma_f32_16x16x32_f16(
                        false, qf[mt][1], false, kf[t][1], (short)0, a, false, false);
                }
            // V fragments are independent of the sigmoid: issue their loads now
            v16h vf[4];
            for (int j = 0; j < 4; j++)
                vf[j] = load_fragB(vb + (size_t)(j * 16 + row16) * SLAB + kk + hi * 8);
            // sigmoid = rcp(1 + exp2(-s)) (QSCALE folded into q), rowsums,
            // stash W tiles in LDS for the D-layout -> A-layout transpose
            for (int mt = 0; mt < 2; mt++) {
                _Float16* wm = ws + mt * 512;
                for (int t = 0; t < 2; t++)
                    for (int r = 0; r < 8; r++) {
                        float e = __builtin_amdgcn_exp2f(-sc[mt][t][r]);
                        float w = __builtin_amdgcn_rcpf(1.0f + e);
                        rowsum[mt][r] += w;
                        wm[(r + hi * 8) * 32 + t * 16 + row16] = (_Float16)w;
                    }
            }
            // reload as A-fragments (same-wave LDS ops stay in order)
            v16h wf[2];
            for (int mt = 0; mt < 2; mt++) {
                const _Float16* p = ws + mt * 512 + row16 * 32 + hi * 8;
                *(uint4*)&wf[mt]       = *(const uint4*)(p);
                *((uint4*)&wf[mt] + 1) = *(const uint4*)(p + 16);
            }
            // O += W @ V over 4 d-subtiles x 2 m-tiles
            for (int mt = 0; mt < 2; mt++)
                for (int j = 0; j < 4; j++)
                    oacc[mt][j] = __builtin_amdgcn_wmma_f32_16x16x32_f16(
                        false, wf[mt], false, vf[j], (short)0, oacc[mt][j], false, false);
        }
        wait_stage();
    }

    // renorm: reduce partial sums across the 16-lane half-wave, then scale
    for (int mt = 0; mt < 2; mt++)
        for (int r = 0; r < 8; r++) {
            float s = rowsum[mt][r];
            s += __shfl_xor(s, 1);
            s += __shfl_xor(s, 2);
            s += __shfl_xor(s, 4);
            s += __shfl_xor(s, 8);
            rowsum[mt][r] = __builtin_amdgcn_rcpf(s + 1e-8f);
        }

    // write attn in [b, l, h*64+d] layout (f16) for the O-projection
    for (int mt = 0; mt < 2; mt++)
        for (int j = 0; j < 4; j++)
            for (int r = 0; r < 8; r++) {
                float v = oacc[mt][j][r] * rowsum[mt][r];
                int m = l0 + mt * 16 + r + hi * 8;
                attn[((size_t)(b * SEQ + m)) * D_MODEL + h * HDIM + j * 16 + row16] =
                    (_Float16)v;
            }
}

// ---------------------------------------------------------------------------
// Host launcher
// ---------------------------------------------------------------------------
extern "C" void kernel_launch(void* const* d_in, const int* in_sizes, int n_in,
                              void* d_out, int out_size, void* d_ws, size_t ws_size,
                              hipStream_t stream) {
    const float* x   = (const float*)d_in[0];
    const float* q_w = (const float*)d_in[1];
    const float* q_b = (const float*)d_in[2];
    const float* k_w = (const float*)d_in[3];
    const float* k_b = (const float*)d_in[4];
    const float* v_w = (const float*)d_in[5];
    const float* v_b = (const float*)d_in[6];
    const float* o_w = (const float*)d_in[7];
    const float* o_b = (const float*)d_in[8];
    const float* qn  = (const float*)d_in[9];
    const float* kn  = (const float*)d_in[10];

    const size_t WBYTES = (size_t)D_MODEL * D_MODEL * sizeof(_Float16); // 2 MB
    const size_t XBYTES = (size_t)MTOT * D_MODEL * sizeof(_Float16);    // 8 MB

    char* ws = (char*)d_ws;
    _Float16* w16q  = (_Float16*)(ws);                     ws += WBYTES;
    _Float16* w16k  = (_Float16*)(ws);                     ws += WBYTES;
    _Float16* w16v  = (_Float16*)(ws);                     ws += WBYTES;
    _Float16* w16o  = (_Float16*)(ws);                     ws += WBYTES;
    _Float16* xq16  = (_Float16*)(ws);                     ws += XBYTES;
    _Float16* xk16  = (_Float16*)(ws);                     ws += XBYTES;
    _Float16* xv16  = (_Float16*)(ws);                     ws += XBYTES;
    _Float16* qpre  = (_Float16*)(ws);                     ws += XBYTES; // [b,h,l,d]
    _Float16* kpre  = (_Float16*)(ws);                     ws += XBYTES; // [b,h,l,d]
    _Float16* q16r  = (_Float16*)(ws);                     ws += XBYTES; // roped
    _Float16* k16r  = (_Float16*)(ws);                     ws += XBYTES; // roped
    _Float16* v16t  = (_Float16*)(ws);                     ws += XBYTES; // [b,h,d,l]
    _Float16* at16  = (_Float16*)(ws);                     ws += XBYTES;

    const int WN = D_MODEL * D_MODEL; // 1M elements per weight

    // 1) weights -> f16
    f32_to_f16_kernel<<<(WN + 255) / 256, 256, 0, stream>>>(q_w, w16q, WN);
    f32_to_f16_kernel<<<(WN + 255) / 256, 256, 0, stream>>>(k_w, w16k, WN);
    f32_to_f16_kernel<<<(WN + 255) / 256, 256, 0, stream>>>(v_w, w16v, WN);
    f32_to_f16_kernel<<<(WN + 255) / 256, 256, 0, stream>>>(o_w, w16o, WN);

    // 2) RMSNorm (q-scale, k-scale) + raw x, all f16
    rmsnorm_prep_kernel<<<MTOT, 256, 0, stream>>>(x, qn, kn, xq16, xk16, xv16);

    // 3) Q/K/V projections; epilogue writes attention-ready layouts directly
    dim3 ggrid(D_MODEL / 256, MTOT / 128);
    gemm_bias_kernel<2><<<ggrid, 256, 0, stream>>>(xq16, w16q, q_b, qpre);
    gemm_bias_kernel<2><<<ggrid, 256, 0, stream>>>(xk16, w16k, k_b, kpre);
    gemm_bias_kernel<3><<<ggrid, 256, 0, stream>>>(xv16, w16v, v_b, v16t);

    // 4) RoPE on [b,h,l,d] (q also folds in QSCALE)
    const int RN = BATCH * NHEADS * SEQ * (HDIM / 2);
    rope_kernel<<<(RN + 255) / 256, 256, 0, stream>>>(qpre, kpre, q16r, k16r);

    // 5) sigmoid attention (WMMA + TDM double-buffered K/V staging)
    attn_sigmoid_kernel<<<dim3(BATCH * NHEADS, SEQ / 128), 128, 0, stream>>>(
        q16r, k16r, v16t, at16);

    // 6) output projection (WMMA GEMM, f32 out -> d_out)
    gemm_bias_kernel<1><<<ggrid, 256, 0, stream>>>(at16, w16o, o_b, d_out);
}
